// ConcatenationEmbeddingModel_68547678044340
// MI455X (gfx1250) — compile-verified, matrix operand order checked by reading, not run on hardware
//
#include <hip/hip_runtime.h>

// ---------------------------------------------------------------------------
// CDNA5 / gfx1250 implementation. wave32, WMMA bf16 16x16x32, fp32 accumulate.
// Round 4: k-loop unrolled by 2 in gemm_accN so the A/B fragment rotation is
// done by register renaming (ping-pong) instead of v_mov chains, eliminating
// the WMMA->VALU WAR hazard NOPs in the steady-state loop.
// ---------------------------------------------------------------------------

typedef __bf16 bf16;
typedef __attribute__((ext_vector_type(16))) __bf16 v16bf;
typedef __attribute__((ext_vector_type(8)))  float  v8f;

#define DEV __device__ __forceinline__

constexpr int BB = 8, STN = 32, TT = 256, IS = 16, ES = 64, HH = 512, HD = 64;
constexpr int G4   = 4 * HH;       // 2048 gates
constexpr int NSEQ = BB * STN;     // 256 sequences
constexpr int M1   = NSEQ * TT;    // 65536 (rows of flattened [seq, t])
constexpr int K1P  = 96;           // 80 (ES+IS) padded to multiple of 32
constexpr int K2   = HH + ES;      // 576 (already multiple of 32)

// ---- WMMA helpers ---------------------------------------------------------
// 16-bit A fragment (16x32, row-major source): lane r=lane&15 is row M=r,
// ko = (lane>=16)?8:0 ; VGPR0..3 hold K = ko+0..7, VGPR4..7 hold K = 16+ko+0..7.
// B fragment for C = A * W^T with row-major W[N,K] is the same pattern with
// lane = column N, so both load 2x 16B contiguous chunks along K.

DEV v16bf make_frag(const bf16* lo, const bf16* hi) {
  v16bf f;
  uint4 a = *(const uint4*)lo;
  uint4 b = *(const uint4*)hi;
  __builtin_memcpy(&f, &a, 16);
  __builtin_memcpy((char*)&f + 16, &b, 16);
  return f;
}

DEV v16bf frag_rm(const bf16* base, long rowStride, int lane, int k0) {
  int r  = lane & 15;
  int ko = (lane >> 4) * 8;
  const bf16* p = base + (long)r * rowStride + k0 + ko;
  return make_frag(p, p + 16);
}

// Gather B fragment from a K-major (column has stride ks) source: B[k][n].
DEV v16bf frag_cg(const bf16* base, int ks, int lane, int k0) {
  int nn = lane & 15;
  int ko = (lane >> 4) * 8;
  v16bf f;
#pragma unroll
  for (int v = 0; v < 4; ++v) {
    int ka = k0 + ko + 2 * v;
    int kb = k0 + 16 + ko + 2 * v;
    f[2 * v]     = base[(long)ka * ks + nn];
    f[2 * v + 1] = base[(long)(ka + 1) * ks + nn];
    f[8 + 2 * v]     = base[(long)kb * ks + nn];
    f[8 + 2 * v + 1] = base[(long)(kb + 1) * ks + nn];
  }
  return f;
}

// C/D 16x16 fp32: VGPR r -> row M = r (lanes 0-15) / 8+r (lanes 16-31), N = lane&15.
DEV void st_f32(float* base, int rs, int lane, v8f acc) {
  int nn = lane & 15, mo = (lane >> 4) * 8;
#pragma unroll
  for (int r = 0; r < 8; ++r) base[(mo + r) * rs + nn] = acc[r];
}
DEV void st_bf16(bf16* base, int rs, int lane, v8f acc) {
  int nn = lane & 15, mo = (lane >> 4) * 8;
#pragma unroll
  for (int r = 0; r < 8; ++r) base[(mo + r) * rs + nn] = (bf16)acc[r];
}

DEV v8f wmma_bf16(v16bf a, v16bf b, v8f c) {
  return __builtin_amdgcn_wmma_f32_16x16x32_bf16(false, a, false, b, (short)0, c,
                                                 false, false);
}

// Pipelined multi-accumulator GEMM along K: NT tiles C[i] += A * B_i with a
// shared A fragment per k-block. B fragments rotate through a double buffer
// so the loads for B_{i+1} (and next-k A) are in flight while WMMA consumes
// B_i -> partial s_wait_loadcnt instead of a full drain before each WMMA.
// The k-loop is unrolled by 2 so the rotation is resolved by register
// renaming (ping-pong) rather than v_mov chains (avoids the WMMA->VALU WAR
// hazard NOPs on the A-fragment registers).
// NOTE: the final rotation issues one dead 32B over-read past the end of the
// A and B streams; callers guarantee those addresses are still mapped
// (adjacent workspace buffer / adjacent LDS region / trailing guard).
template <int KTOT, int NT>
DEV void gemm_accN(v8f* acc, const bf16* Abase, long aStride,
                   const bf16* Wbase, long wStride, int lane) {
  v16bf a = frag_rm(Abase, aStride, lane, 0);
  v16bf b = frag_rm(Wbase, wStride, lane, 0);
#pragma unroll 2
  for (int k = 0; k < KTOT; k += 32) {
    __builtin_prefetch((const char*)(Wbase + k + 64), 0, 3); // stream hint
    v16bf aN = frag_rm(Abase, aStride, lane, k + 32); // next-k A (dead on last iter)
#pragma unroll
    for (int i = 0; i < NT; ++i) {
      v16bf bN = (i + 1 < NT)
                     ? frag_rm(Wbase + (long)(i + 1) * 16 * wStride, wStride, lane, k)
                     : frag_rm(Wbase, wStride, lane, k + 32); // dead on last iter
      acc[i] = wmma_bf16(a, b, acc[i]);
      b = bN;
    }
    a = aN;
  }
}

DEV float sigmoidf(float x) { return 1.f / (1.f + __expf(-x)); }
DEV float tanh_fast(float x) {
  x = fminf(fmaxf(x, -15.f), 15.f);
  float e = __expf(2.f * x);
  return (e - 1.f) / (e + 1.f);
}
DEV unsigned pack_bf16(float a, float b) {
  bf16 x = (bf16)a, y = (bf16)b;
  unsigned short ux, uy;
  __builtin_memcpy(&ux, &x, 2);
  __builtin_memcpy(&uy, &y, 2);
  return (unsigned)ux | ((unsigned)uy << 16);
}

// ---- small prep kernels ---------------------------------------------------

__global__ void k_convert_pad(const float* __restrict__ src, bf16* __restrict__ dst,
                              int N, int K, int Kp) {
  long idx = (long)blockIdx.x * blockDim.x + threadIdx.x;
  if (idx >= (long)N * Kp) return;
  int  k = (int)(idx % Kp);
  long r = idx / Kp;
  dst[idx] = (k < K) ? (bf16)src[r * K + k] : (bf16)0.f;
}

__global__ void k_fill_x1(const int* __restrict__ e, const float* __restrict__ x,
                          const float* __restrict__ emb1, bf16* __restrict__ X1) {
  long idx = (long)blockIdx.x * blockDim.x + threadIdx.x;
  if (idx >= (long)M1 * K1P) return;
  int  k = (int)(idx % K1P);
  long r = idx / K1P;
  float v;
  if (k < ES)            v = emb1[(long)e[r] * ES + k];
  else if (k < ES + IS)  v = x[r * IS + (k - ES)];
  else                   v = 0.f;
  X1[idx] = (bf16)v;
}

__global__ void k_fill_x2emb(const int* __restrict__ e, const float* __restrict__ emb2,
                             bf16* __restrict__ X2) {
  long idx = (long)blockIdx.x * blockDim.x + threadIdx.x;
  if (idx >= (long)M1 * ES) return;
  int  k = (int)(idx % ES);
  long r = idx / ES;
  X2[r * K2 + k] = (bf16)emb2[(long)e[r] * ES + k];
}

// ---- fused LSTM scan ------------------------------------------------------
// One block = 16 sequences (one WMMA M-tile), 256 threads = 8 wave32.
// Per step: g[16,2048] = b + x_t @ Wih^T + h @ Whh^T. Each wave owns 16
// N-tiles, processed as 2 groups of 8 live accumulators fed by the pipelined
// gemm_accN helper. c stays in VGPRs, h in LDS.

template <int KIN>
__global__ void k_lstm_scan(const bf16* __restrict__ X,
                            const bf16* __restrict__ Wih, const bf16* __restrict__ Whh,
                            const float* __restrict__ bih, const float* __restrict__ bhh,
                            const float* __restrict__ linw, const float* __restrict__ linb,
                            const int* __restrict__ e, const float* __restrict__ attnEmb,
                            bf16* __restrict__ Yout, float* __restrict__ target) {
  extern __shared__ char smem[];
  bf16*  hL  = (bf16*)smem;                                // [16][512]  16KB
  bf16*  gL  = (bf16*)(smem + 16 * HH * 2);                // [16][2048] 64KB
  float* red = (float*)(smem + 16 * HH * 2 + 16 * G4 * 2); // [16][16]   1KB

  const int tid  = threadIdx.x;
  const int w    = tid >> 5, lane = tid & 31;
  const int s0   = blockIdx.x * 16;

  for (int i = tid; i < 16 * HH; i += 256) hL[i] = (bf16)0.f;

  float creg[32];
#pragma unroll
  for (int j = 0; j < 32; ++j) creg[j] = 0.f;

  // hoist per-tile gate bias (i,f,g,o biases at this lane's column) out of t-loop
  float bvr[16];
#pragma unroll
  for (int i = 0; i < 16; ++i) {
    int nc = (w * 16 + i) * 16 + (lane & 15);
    bvr[i] = bih[nc] + bhh[nc];
  }

  const int row   = tid >> 4;        // 0..15 : sequence within tile
  const int cb    = (tid & 15) * 32; // 32 hidden cells per thread
  const int s     = s0 + row;
  const int bidx  = s >> 5, stidx = s & 31;

  __syncthreads();

  for (int t = 0; t < TT; ++t) {
    // ---------------- gate GEMM ----------------
    const bf16* Xrow0 = X + ((long)s0 * TT + t) * KIN; // A-tile row stride = TT*KIN
    for (int g = 0; g < 2; ++g) {
      const int nb = (w * 16 + g * 8) * 16; // first N-column of this group
      v8f acc[8];
#pragma unroll
      for (int i = 0; i < 8; ++i) {
        float bv = bvr[g * 8 + i];
#pragma unroll
        for (int r = 0; r < 8; ++r) acc[i][r] = bv;
      }
      // input contribution: pipelined, A fragment from global X
      gemm_accN<KIN, 8>(acc, Xrow0, (long)TT * KIN, Wih + (long)nb * KIN, KIN, lane);
      // recurrent contribution: pipelined, A fragment from LDS h
      gemm_accN<HH, 8>(acc, hL, HH, Whh + (long)nb * HH, HH, lane);
#pragma unroll
      for (int i = 0; i < 8; ++i) st_bf16(gL + nb + i * 16, G4, lane, acc[i]);
    }
    __syncthreads();

    // ---------------- cell update ----------------
    int   eidx = (Yout != nullptr) ? e[(long)s * TT + t] : 0;
    float partial = 0.f;
#pragma unroll
    for (int jj = 0; jj < 32; jj += 2) {
      int   j   = cb + jj;
      float h01[2];
#pragma unroll
      for (int u = 0; u < 2; ++u) {
        float gi = (float)gL[row * G4 + j + u];
        float gf = (float)gL[row * G4 + j + u + HH];
        float gg = (float)gL[row * G4 + j + u + 2 * HH];
        float go = (float)gL[row * G4 + j + u + 3 * HH];
        float c  = sigmoidf(gf) * creg[jj + u] + sigmoidf(gi) * tanh_fast(gg);
        creg[jj + u] = c;
        h01[u]   = sigmoidf(go) * tanh_fast(c);
        partial += h01[u] * linw[j + u];
      }
      *(unsigned*)(hL + row * HH + j) = pack_bf16(h01[0], h01[1]);
      if (Yout) { // LSTM1: write [b, t, st, h] with attention embedding added
        float y0 = h01[0] + attnEmb[(long)eidx * HH + j];
        float y1 = h01[1] + attnEmb[(long)eidx * HH + j + 1];
        unsigned* dst =
            (unsigned*)(Yout + (((long)bidx * TT + t) * STN + stidx) * HH + j);
        *dst = pack_bf16(y0, y1);
      }
    }
    red[row * 16 + (tid & 15)] = partial;
    __syncthreads();
    if ((tid & 15) == 0) {
      float sum = linb[0];
#pragma unroll
      for (int q = 0; q < 16; ++q) sum += red[row * 16 + q];
      target[(long)s * TT + t] = sum;
    }
    __syncthreads(); // h fully written before next step's GEMM reads
  }
}

// ---- fused station attention + layernorm ----------------------------------
// One block per (b, t): 32 stations x 512 features. qkv tile held in LDS,
// wave w owns head w for scores/softmax/PV. Output written into X2[..., 64:].

__global__ void k_attn(const bf16* __restrict__ Y,
                       const bf16* __restrict__ Wqkv, const float* __restrict__ bqkv,
                       const bf16* __restrict__ Wo,   const float* __restrict__ bo,
                       const float* __restrict__ lng, const float* __restrict__ lnb,
                       bf16* __restrict__ X2) {
  extern __shared__ char smem[];
  bf16*  yL  = (bf16*)smem;                                  // [32][512]   32KB
  bf16*  qkv = (bf16*)(smem + 32 * HH * 2);                  // [32][1536]  96KB
  float* sc  = (float*)(smem + 32 * HH * 2 + 32 * 3 * HH * 2); // 8 x [32][34] 34KB
  bf16*  pL  = (bf16*)((char*)sc + 8 * 32 * 34 * 4);         // 8 x [32][32] 16KB
  bf16*  rL  = (bf16*)sc; // aliases sc after PV barrier: [32][512] bf16

  const int tid = threadIdx.x;
  const int w   = tid >> 5, lane = tid & 31;
  const int n   = blockIdx.x;       // b*T + t
  const int b   = n / TT, t = n % TT;

  { // load y tile [32][512] bf16 (2048 x uint4)
    const uint4* src = (const uint4*)(Y + (long)n * 32 * HH);
    uint4*       dst = (uint4*)yL;
    for (int i = tid; i < 32 * HH / 8; i += 256) dst[i] = src[i];
  }
  __syncthreads();

  // qkv = y @ Wqkv^T + b : per m-half, 12 N-tiles per wave (3 groups of 4)
  for (int half = 0; half < 2; ++half) {
    int m0 = half * 16;
    for (int grp = 0; grp < 3; ++grp) {
      int nbase = (w * 12 + grp * 4) * 16; // 96 N-tiles total: wave w owns 12
      v8f acc[4];
#pragma unroll
      for (int i = 0; i < 4; ++i) {
        float bv = bqkv[nbase + i * 16 + (lane & 15)];
#pragma unroll
        for (int r = 0; r < 8; ++r) acc[i][r] = bv;
      }
      gemm_accN<HH, 4>(acc, yL + m0 * HH, HH, Wqkv + (long)nbase * HH, HH, lane);
#pragma unroll
      for (int i = 0; i < 4; ++i)
        st_bf16(qkv + m0 * 3 * HH + nbase + i * 16, 3 * HH, lane, acc[i]);
    }
  }
  __syncthreads();

  { // wave w handles head w entirely
    const bf16* qb  = qkv + w * HD;
    const bf16* kb  = qkv + HH + w * HD;
    float*      sch = sc + w * 32 * 34;

    // scores = Q K^T : per m-half, 2 N-tiles pipelined (B tiles = K rows)
    for (int half = 0; half < 2; ++half) {
      int m0 = half * 16;
      v8f acc[2];
#pragma unroll
      for (int i = 0; i < 2; ++i)
#pragma unroll
        for (int r = 0; r < 8; ++r) acc[i][r] = 0.f;
      gemm_accN<HD, 2>(acc, qb + m0 * 3 * HH, 3 * HH, kb, 3 * HH, lane);
      st_f32(sch + m0 * 34 + 0, 34, lane, acc[0]);
      st_f32(sch + m0 * 34 + 16, 34, lane, acc[1]);
    }

    // softmax: lane = row (scale 1/sqrt(64) = 0.125)
    float* srow = sch + lane * 34;
    float  mx = -1e30f;
    for (int c = 0; c < 32; ++c) mx = fmaxf(mx, srow[c]);
    float sum = 0.f;
    for (int c = 0; c < 32; ++c) {
      float p = __expf((srow[c] - mx) * 0.125f);
      sum += p;
      srow[c] = p;
    }
    float inv = 1.f / sum;
    bf16* prow = pL + w * 32 * 32 + lane * 32;
    for (int c = 0; c < 32; c += 2)
      *(unsigned*)(prow + c) = pack_bf16(srow[c] * inv, srow[c + 1] * inv);

    // O = P V (K=32 -> one WMMA per tile); overwrite this head's q slot
    const bf16* vb = qkv + 2 * HH + w * HD;
    bf16*       ob = qkv + w * HD;
    for (int tIdx = 0; tIdx < 8; ++tIdx) {
      int m0 = (tIdx & 1) * 16, n0 = (tIdx >> 1) * 16;
      v8f acc;
#pragma unroll
      for (int r = 0; r < 8; ++r) acc[r] = 0.f;
      v16bf a = frag_rm(pL + w * 32 * 32 + m0 * 32, 32, lane, 0);
      v16bf bb = frag_cg(vb + n0, 3 * HH, lane, 0);
      acc = wmma_bf16(a, bb, acc);
      st_bf16(ob + m0 * 3 * HH + n0, 3 * HH, lane, acc);
    }
  }
  __syncthreads();

  // out_proj + residual -> rL (bf16, aliases sc region); 4 live accs, pipelined
  for (int half = 0; half < 2; ++half) {
    int m0 = half * 16;
    int nbase = w * 4 * 16; // 32 N-tiles: wave w owns 4
    v8f acc[4];
#pragma unroll
    for (int i = 0; i < 4; ++i) {
      float bv = bo[nbase + i * 16 + (lane & 15)];
#pragma unroll
      for (int r = 0; r < 8; ++r) acc[i][r] = bv;
    }
    gemm_accN<HH, 4>(acc, qkv + m0 * 3 * HH, 3 * HH, Wo + (long)nbase * HH, HH, lane);
    int nn = lane & 15, mo = (lane >> 4) * 8;
#pragma unroll
    for (int i = 0; i < 4; ++i) {
      int nc = nbase + i * 16 + nn;
#pragma unroll
      for (int r = 0; r < 8; ++r) {
        float v = acc[i][r] + (float)yL[(m0 + mo + r) * HH + nc];
        rL[(m0 + mo + r) * HH + nc] = (bf16)v;
      }
    }
  }
  __syncthreads();

  // layernorm per station row, write into X2[:, 64:576]
  if (tid < 32) {
    int   st = tid;
    float mean = 0.f, var = 0.f;
    for (int c = 0; c < HH; ++c) mean += (float)rL[st * HH + c];
    mean *= (1.f / HH);
    for (int c = 0; c < HH; ++c) {
      float v = (float)rL[st * HH + c] - mean;
      var += v * v;
    }
    var *= (1.f / HH);
    float inv = rsqrtf(var + 1e-5f);
    long  sidx = (long)b * STN + st;
    bf16* dst  = X2 + (sidx * TT + t) * (long)K2 + ES;
    for (int c = 0; c < HH; ++c) {
      float v = ((float)rL[st * HH + c] - mean) * inv * lng[c] + lnb[c];
      dst[c] = (bf16)v;
    }
  }
}

// ---------------------------------------------------------------------------

extern "C" void kernel_launch(void* const* d_in, const int* in_sizes, int n_in,
                              void* d_out, int out_size, void* d_ws, size_t ws_size,
                              hipStream_t stream) {
  const int*   e    = (const int*)d_in[0];
  const float* x    = (const float*)d_in[1];
  const float* emb1 = (const float*)d_in[2];
  const float* Wih1 = (const float*)d_in[3];
  const float* Whh1 = (const float*)d_in[4];
  const float* bih1 = (const float*)d_in[5];
  const float* bhh1 = (const float*)d_in[6];
  const float* l1w  = (const float*)d_in[7];
  const float* l1b  = (const float*)d_in[8];
  const float* aemb = (const float*)d_in[9];
  const float* Wqkv = (const float*)d_in[10];
  const float* bqkv = (const float*)d_in[11];
  const float* Wo   = (const float*)d_in[12];
  const float* bo   = (const float*)d_in[13];
  const float* lng  = (const float*)d_in[14];
  const float* lnb  = (const float*)d_in[15];
  const float* emb2 = (const float*)d_in[16];
  const float* Wih2 = (const float*)d_in[17];
  const float* Whh2 = (const float*)d_in[18];
  const float* bih2 = (const float*)d_in[19];
  const float* bhh2 = (const float*)d_in[20];
  const float* l2w  = (const float*)d_in[21];
  const float* l2b  = (const float*)d_in[22];

  float* target1 = (float*)d_out;
  float* target2 = target1 + M1;

  char* wsp = (char*)d_ws;
  auto carve = [&](size_t bytes) -> char* {
    char* p = wsp;
    wsp += (bytes + 255) & ~(size_t)255;
    return p;
  };
  bf16* Wih1b = (bf16*)carve((size_t)G4 * K1P * 2);
  bf16* Whh1b = (bf16*)carve((size_t)G4 * HH * 2);
  bf16* Wih2b = (bf16*)carve((size_t)G4 * K2 * 2);
  bf16* Whh2b = (bf16*)carve((size_t)G4 * HH * 2);
  bf16* Wqkvb = (bf16*)carve((size_t)3 * HH * HH * 2);
  bf16* Wob   = (bf16*)carve((size_t)HH * HH * 2);
  bf16* X1    = (bf16*)carve((size_t)M1 * K1P * 2);
  bf16* Yb    = (bf16*)carve((size_t)M1 * HH * 2);
  bf16* X2    = (bf16*)carve((size_t)M1 * K2 * 2);
  (void)carve(4096); // guard region: absorbs the pipeline's trailing over-reads

  auto blocks = [](long nel) { return dim3((unsigned)((nel + 255) / 256)); };

  k_convert_pad<<<blocks((long)G4 * K1P), 256, 0, stream>>>(Wih1, Wih1b, G4, ES + IS, K1P);
  k_convert_pad<<<blocks((long)G4 * HH), 256, 0, stream>>>(Whh1, Whh1b, G4, HH, HH);
  k_convert_pad<<<blocks((long)G4 * K2), 256, 0, stream>>>(Wih2, Wih2b, G4, K2, K2);
  k_convert_pad<<<blocks((long)G4 * HH), 256, 0, stream>>>(Whh2, Whh2b, G4, HH, HH);
  k_convert_pad<<<blocks((long)3 * HH * HH), 256, 0, stream>>>(Wqkv, Wqkvb, 3 * HH, HH, HH);
  k_convert_pad<<<blocks((long)HH * HH), 256, 0, stream>>>(Wo, Wob, HH, HH, HH);

  k_fill_x1<<<blocks((long)M1 * K1P), 256, 0, stream>>>(e, x, emb1, X1);

  size_t lstmLds = 16 * HH * 2 + 16 * G4 * 2 + 16 * 16 * 4; // 81.0 KB
  k_lstm_scan<K1P><<<dim3(NSEQ / 16), 256, lstmLds, stream>>>(
      X1, Wih1b, Whh1b, bih1, bhh1, l1w, l1b, e, aemb, Yb, target1);

  k_fill_x2emb<<<blocks((long)M1 * ES), 256, 0, stream>>>(e, emb2, X2);

  size_t attnLds = 32 * HH * 2 + 32 * 3 * HH * 2 + 8 * 32 * 34 * 4 + 8 * 32 * 32 * 2; // ~178 KB (<320KB/WGP)
  k_attn<<<dim3(BB * TT), 256, attnLds, stream>>>(
      Yb, Wqkvb, bqkv, Wob, bo, lng, lnb, X2);

  k_lstm_scan<K2><<<dim3(NSEQ / 16), 256, lstmLds, stream>>>(
      X2, Wih2b, Whh2b, bih2, bhh2, l2w, l2b, nullptr, nullptr, nullptr, target2);
}
